// Net_3350074491288
// MI455X (gfx1250) — matrix operand order repaired and checked by reading, not compiled, not run on hardware
//
#include <hip/hip_runtime.h>
#include <math.h>

typedef __attribute__((ext_vector_type(16))) _Float16 v16h;
typedef __attribute__((ext_vector_type(8)))  _Float16 v8h;
typedef __attribute__((ext_vector_type(8)))  float    v8f;

#define N_NODES 50000
#define E_EDGES 800000
#define ETOT    (E_EDGES + N_NODES)   // 850000, self-loops appended
#define F_IN    128
#define G_GRAPHS 500
#define H1 4
#define C1 32
#define D1 128          // H1*C1
#define C2 8
#define MTILES (N_NODES/16)  // 3125 exactly

// ---------------- utility ----------------
__global__ void k_fill(float* __restrict__ p, float v, int n) {
    int i = blockIdx.x * blockDim.x + threadIdx.x;
    if (i < n) p[i] = v;
}

__device__ __forceinline__ void atomicMaxF(float* a, float v) {
    // classic monotone bit trick; target initialized to -inf bits
    if (v >= 0.f) atomicMax((int*)a, __float_as_int(v));
    else          atomicMin((unsigned int*)a, __float_as_uint(v));
}

__device__ __forceinline__ int edge_src(const int* __restrict__ ei, int e) {
    return (e < E_EDGES) ? ei[e] : (e - E_EDGES);
}
__device__ __forceinline__ int edge_dst(const int* __restrict__ ei, int e) {
    return (e < E_EDGES) ? ei[E_EDGES + e] : (e - E_EDGES);
}

// ---------------- BatchNorm ----------------
__global__ void k_bn_sums(const float* __restrict__ x, float* __restrict__ s, float* __restrict__ s2) {
    int j = threadIdx.x;                       // feature 0..127
    int rows = (N_NODES + gridDim.x - 1) / gridDim.x;
    int r0 = blockIdx.x * rows;
    int r1 = min(r0 + rows, N_NODES);
    float a = 0.f, b = 0.f;
    for (int r = r0; r < r1; ++r) {
        float v = x[r * F_IN + j];
        a += v; b += v * v;
    }
    atomicAdd(&s[j], a);
    atomicAdd(&s2[j], b);
}

__global__ void k_bn_finalize(const float* __restrict__ s, const float* __restrict__ s2,
                              const float* __restrict__ gamma, const float* __restrict__ beta,
                              float* __restrict__ scale, float* __restrict__ shift) {
    int j = threadIdx.x;
    float mean = s[j] * (1.f / (float)N_NODES);
    float var  = s2[j] * (1.f / (float)N_NODES) - mean * mean;
    float inv  = rsqrtf(var + 1e-5f);
    float sc   = inv * gamma[j];
    scale[j] = sc;
    shift[j] = beta[j] - mean * sc;
}

__global__ void k_bn_apply(const float* __restrict__ x, const float* __restrict__ scale,
                           const float* __restrict__ shift, _Float16* __restrict__ xh) {
    int i = blockIdx.x * blockDim.x + threadIdx.x;
    if (i >= N_NODES * F_IN) return;
    int j = i & 127;
    xh[i] = (_Float16)(x[i] * scale[j] + shift[j]);
}

// ---------------- weight packing into WMMA B-fragment layout ----------------
// B (KxN, K=32 per step): lanes 0-15 -> cols, K 0..7,16..23; lanes 16-31 -> K 8..15,24..31.
// Fragment for (ntile,kstep): 32 lanes x 16 halves, contiguous per lane.
__global__ void k_pack_w1(const float* __restrict__ W, _Float16* __restrict__ wp) {
    int i = blockIdx.x * blockDim.x + threadIdx.x;      // 8*4*32*16 = 16384
    if (i >= 8 * 4 * 32 * 16) return;
    int h = i & 15, lane = (i >> 4) & 31, k = (i >> 9) & 3, nt = i >> 11;
    int col = nt * 16 + (lane & 15);
    int kk  = k * 32 + (lane >> 4) * 8 + (h & 7) + (h >> 3) * 16;
    wp[i] = (_Float16)W[kk * D1 + col];
}

__global__ void k_pack_w2(const float* __restrict__ W, _Float16* __restrict__ wp) {
    int i = blockIdx.x * blockDim.x + threadIdx.x;      // 4*32*16 = 2048
    if (i >= 4 * 32 * 16) return;
    int h = i & 15, lane = (i >> 4) & 31, k = i >> 9;
    int col = lane & 15;
    int kk  = k * 32 + (lane >> 4) * 8 + (h & 7) + (h >> 3) * 16;
    wp[i] = (col < C2) ? (_Float16)W[kk * C2 + col] : (_Float16)0.0f;
}

// ---------------- WMMA GEMM: [N_NODES,128] x [128, 16*ntiles] ----------------
// one wave per 16x16 output tile; A row-major f16, B pre-packed fragments.
__global__ void k_gemm_wmma(const _Float16* __restrict__ A, const _Float16* __restrict__ Bp,
                            float* __restrict__ Cout, int ntiles, int ldc, int storeCols) {
    int lane = threadIdx.x & 31;
    int gw   = blockIdx.x * (blockDim.x >> 5) + (threadIdx.x >> 5);
    int mtile = gw / ntiles;
    int ntile = gw % ntiles;
    if (mtile >= MTILES) return;

    int row = mtile * 16 + (lane & 15);
    int klo = (lane >> 4) * 8;           // dense 16-bit A 16x32 layout
    v8f c = {};
    #pragma unroll
    for (int k = 0; k < 4; ++k) {
        const _Float16* ap = A + row * F_IN + k * 32 + klo;
        v8h alo = *(const v8h*)ap;             // K = klo..klo+7
        v8h ahi = *(const v8h*)(ap + 16);      // K = klo+16..klo+23
        v16h a;
        #pragma unroll
        for (int t = 0; t < 8; ++t) { a[t] = alo[t]; a[8 + t] = ahi[t]; }
        v16h b = *(const v16h*)(Bp + (((ntile * 4 + k) * 32 + lane) << 4));
        c = __builtin_amdgcn_wmma_f32_16x16x32_f16(false, a, false, b, (short)0, c, false, false);
    }
    int col   = ntile * 16 + (lane & 15);
    int rbase = mtile * 16 + 8 * (lane >> 4);
    if (col < storeCols) {
        #pragma unroll
        for (int r = 0; r < 8; ++r)
            Cout[(rbase + r) * ldc + col] = c[r];
    }
}

// ---------------- attention coefficients ----------------
__global__ void k_att1(const float* __restrict__ h, const float* __restrict__ as,
                       const float* __restrict__ ad, float* __restrict__ al, float* __restrict__ ar) {
    int lane = threadIdx.x & 31;
    int n = blockIdx.x * (blockDim.x >> 5) + (threadIdx.x >> 5);
    if (n >= N_NODES) return;
    float sa[H1], sd[H1];
    #pragma unroll
    for (int hh = 0; hh < H1; ++hh) {
        float v = h[n * D1 + hh * C1 + lane];
        sa[hh] = v * as[hh * C1 + lane];
        sd[hh] = v * ad[hh * C1 + lane];
    }
    #pragma unroll
    for (int hh = 0; hh < H1; ++hh) {
        for (int off = 16; off; off >>= 1) {
            sa[hh] += __shfl_xor(sa[hh], off, 32);
            sd[hh] += __shfl_xor(sd[hh], off, 32);
        }
    }
    if (lane == 0) {
        #pragma unroll
        for (int hh = 0; hh < H1; ++hh) { al[n * H1 + hh] = sa[hh]; ar[n * H1 + hh] = sd[hh]; }
    }
}

__global__ void k_att2(const float* __restrict__ h, const float* __restrict__ as,
                       const float* __restrict__ ad, float* __restrict__ al, float* __restrict__ ar) {
    int n = blockIdx.x * blockDim.x + threadIdx.x;
    if (n >= N_NODES) return;
    float sa = 0.f, sd = 0.f;
    #pragma unroll
    for (int c = 0; c < C2; ++c) {
        float v = h[n * C2 + c];
        sa += v * as[c]; sd += v * ad[c];
    }
    al[n] = sa; ar[n] = sd;
}

// ---------------- GAT layer 1 edge passes ----------------
__global__ void k_edge_logits1(const int* __restrict__ ei, const float* __restrict__ al,
                               const float* __restrict__ ar, float* __restrict__ m, float* __restrict__ elog) {
    int i = blockIdx.x * blockDim.x + threadIdx.x;
    if (i >= ETOT * H1) return;
    int e = i >> 2, hh = i & 3;
    int s = edge_src(ei, e), d = edge_dst(ei, e);
    float v = al[s * H1 + hh] + ar[d * H1 + hh];
    v = v > 0.f ? v : 0.2f * v;     // LeakyReLU
    elog[i] = v;
    atomicMaxF(&m[d * H1 + hh], v);
}

__global__ void k_edge_exp1(const int* __restrict__ ei, const float* __restrict__ m,
                            float* __restrict__ elog, float* __restrict__ den) {
    int i = blockIdx.x * blockDim.x + threadIdx.x;
    if (i >= ETOT * H1) return;
    int e = i >> 2, hh = i & 3;
    int d = edge_dst(ei, e);
    float ex = __expf(elog[i] - m[d * H1 + hh]);
    elog[i] = ex;
    atomicAdd(&den[d * H1 + hh], ex);
}

__global__ void k_edge_aggr1(const int* __restrict__ ei, const float* __restrict__ ex,
                             const float* __restrict__ den, const float* __restrict__ h,
                             float* __restrict__ out) {
    int lane = threadIdx.x & 31;
    int e = blockIdx.x * (blockDim.x >> 5) + (threadIdx.x >> 5);
    if (e >= ETOT) return;
    int s = edge_src(ei, e), d = edge_dst(ei, e);
    #pragma unroll
    for (int hh = 0; hh < H1; ++hh) {
        float alpha = ex[e * H1 + hh] / (den[d * H1 + hh] + 1e-16f);
        int c = hh * C1 + lane;
        atomicAdd(&out[d * D1 + c], h[s * D1 + c] * alpha);
    }
}

// ---------------- ELU + bias + f32->f16 ----------------
__global__ void k_elu_f16(const float* __restrict__ agg, const float* __restrict__ b1,
                          _Float16* __restrict__ xh) {
    int i = blockIdx.x * blockDim.x + threadIdx.x;
    if (i >= N_NODES * D1) return;
    float v = agg[i] + b1[i & 127];
    v = v > 0.f ? v : (__expf(v) - 1.f);
    xh[i] = (_Float16)v;
}

// ---------------- GAT layer 2 edge passes (H=1) ----------------
__global__ void k_edge_logits2(const int* __restrict__ ei, const float* __restrict__ al,
                               const float* __restrict__ ar, float* __restrict__ m, float* __restrict__ elog) {
    int e = blockIdx.x * blockDim.x + threadIdx.x;
    if (e >= ETOT) return;
    int s = edge_src(ei, e), d = edge_dst(ei, e);
    float v = al[s] + ar[d];
    v = v > 0.f ? v : 0.2f * v;
    elog[e] = v;
    atomicMaxF(&m[d], v);
}

__global__ void k_edge_exp2(const int* __restrict__ ei, const float* __restrict__ m,
                            float* __restrict__ elog, float* __restrict__ den) {
    int e = blockIdx.x * blockDim.x + threadIdx.x;
    if (e >= ETOT) return;
    int d = edge_dst(ei, e);
    float ex = __expf(elog[e] - m[d]);
    elog[e] = ex;
    atomicAdd(&den[d], ex);
}

__global__ void k_edge_aggr2(const int* __restrict__ ei, const float* __restrict__ ex,
                             const float* __restrict__ den, const float* __restrict__ h,
                             float* __restrict__ out) {
    int i = blockIdx.x * blockDim.x + threadIdx.x;
    if (i >= ETOT * C2) return;
    int e = i >> 3, c = i & 7;
    int s = edge_src(ei, e), d = edge_dst(ei, e);
    float alpha = ex[e] / (den[d] + 1e-16f);
    atomicAdd(&out[d * C2 + c], h[s * C2 + c] * alpha);
}

// ---------------- pooling + log_softmax ----------------
__global__ void k_pool(const int* __restrict__ batch, const float* __restrict__ o2,
                       const float* __restrict__ b2, float* __restrict__ pooled, float* __restrict__ cnt) {
    int i = blockIdx.x * blockDim.x + threadIdx.x;
    if (i >= N_NODES * C2) return;
    int n = i >> 3, c = i & 7;
    int g = batch[n];
    atomicAdd(&pooled[g * C2 + c], o2[i] + b2[c]);
    if (c == 0) atomicAdd(&cnt[g], 1.0f);
}

__global__ void k_logsoftmax(const float* __restrict__ pooled, const float* __restrict__ cnt,
                             float* __restrict__ y) {
    int g = blockIdx.x * blockDim.x + threadIdx.x;
    if (g >= G_GRAPHS) return;
    float inv = 1.f / fmaxf(cnt[g], 1.0f);
    float v[C2];
    float mx = -1e30f;
    #pragma unroll
    for (int c = 0; c < C2; ++c) { v[c] = pooled[g * C2 + c] * inv; mx = fmaxf(mx, v[c]); }
    float s = 0.f;
    #pragma unroll
    for (int c = 0; c < C2; ++c) s += __expf(v[c] - mx);
    float l = logf(s);
    #pragma unroll
    for (int c = 0; c < C2; ++c) y[g * C2 + c] = v[c] - mx - l;
}

// ---------------- launcher ----------------
static inline size_t alignup(size_t x) { return (x + 255) & ~(size_t)255; }

extern "C" void kernel_launch(void* const* d_in, const int* in_sizes, int n_in,
                              void* d_out, int out_size, void* d_ws, size_t ws_size,
                              hipStream_t stream) {
    const float* x        = (const float*)d_in[0];
    const float* bn_gamma = (const float*)d_in[1];
    const float* bn_beta  = (const float*)d_in[2];
    const float* W1       = (const float*)d_in[3];
    const float* att_s1   = (const float*)d_in[4];
    const float* att_d1   = (const float*)d_in[5];
    const float* b1       = (const float*)d_in[6];
    const float* W2       = (const float*)d_in[7];
    const float* att_s2   = (const float*)d_in[8];
    const float* att_d2   = (const float*)d_in[9];
    const float* b2       = (const float*)d_in[10];
    const int*   ei       = (const int*)d_in[11];
    const int*   batch    = (const int*)d_in[12];
    float* y = (float*)d_out;

    // workspace carve-out
    char* ws = (char*)d_ws;
    size_t off = 0;
    auto take = [&](size_t bytes) { char* p = ws + off; off = alignup(off + bytes); return (void*)p; };

    _Float16* xnh   = (_Float16*)take((size_t)N_NODES * F_IN * 2);  // also reused as x1h
    _Float16* w1p   = (_Float16*)take(8 * 4 * 32 * 16 * 2);
    _Float16* w2p   = (_Float16*)take(4 * 32 * 16 * 2);
    float* h1    = (float*)take((size_t)N_NODES * D1 * 4);
    float* out1  = (float*)take((size_t)N_NODES * D1 * 4);
    float* al1   = (float*)take((size_t)N_NODES * H1 * 4);
    float* ar1   = (float*)take((size_t)N_NODES * H1 * 4);
    float* m1    = (float*)take((size_t)N_NODES * H1 * 4);
    float* den1  = (float*)take((size_t)N_NODES * H1 * 4);
    float* ex1   = (float*)take((size_t)ETOT * H1 * 4);
    float* h2    = (float*)take((size_t)N_NODES * C2 * 4);
    float* out2  = (float*)take((size_t)N_NODES * C2 * 4);
    float* al2   = (float*)take((size_t)N_NODES * 4);
    float* ar2   = (float*)take((size_t)N_NODES * 4);
    float* m2    = (float*)take((size_t)N_NODES * 4);
    float* den2  = (float*)take((size_t)N_NODES * 4);
    float* ex2   = (float*)take((size_t)ETOT * 4);
    float* pooled = (float*)take((size_t)G_GRAPHS * C2 * 4);
    float* cnt    = (float*)take((size_t)G_GRAPHS * 4);
    float* bns    = (float*)take(F_IN * 4);
    float* bns2   = (float*)take(F_IN * 4);
    float* bnsc   = (float*)take(F_IN * 4);
    float* bnsh   = (float*)take(F_IN * 4);
    (void)ws_size; (void)n_in; (void)in_sizes; (void)out_size;

    const int BT = 256;
    auto grid = [](long n, int bt) { return (unsigned)((n + bt - 1) / bt); };
    const float NEGINF = -__builtin_huge_valf();

    // ---- per-call init (no cross-call state) ----
    k_fill<<<grid(F_IN, BT), BT, 0, stream>>>(bns, 0.f, F_IN);
    k_fill<<<grid(F_IN, BT), BT, 0, stream>>>(bns2, 0.f, F_IN);
    k_fill<<<grid((long)N_NODES * D1, BT), BT, 0, stream>>>(out1, 0.f, N_NODES * D1);
    k_fill<<<grid((long)N_NODES * H1, BT), BT, 0, stream>>>(den1, 0.f, N_NODES * H1);
    k_fill<<<grid((long)N_NODES * H1, BT), BT, 0, stream>>>(m1, NEGINF, N_NODES * H1);
    k_fill<<<grid((long)N_NODES * C2, BT), BT, 0, stream>>>(out2, 0.f, N_NODES * C2);
    k_fill<<<grid(N_NODES, BT), BT, 0, stream>>>(den2, 0.f, N_NODES);
    k_fill<<<grid(N_NODES, BT), BT, 0, stream>>>(m2, NEGINF, N_NODES);
    k_fill<<<grid(G_GRAPHS * C2, BT), BT, 0, stream>>>(pooled, 0.f, G_GRAPHS * C2);
    k_fill<<<grid(G_GRAPHS, BT), BT, 0, stream>>>(cnt, 0.f, G_GRAPHS);

    // ---- BatchNorm ----
    k_bn_sums<<<200, F_IN, 0, stream>>>(x, bns, bns2);
    k_bn_finalize<<<1, F_IN, 0, stream>>>(bns, bns2, bn_gamma, bn_beta, bnsc, bnsh);
    k_bn_apply<<<grid((long)N_NODES * F_IN, BT), BT, 0, stream>>>(x, bnsc, bnsh, xnh);

    // ---- weight packing ----
    k_pack_w1<<<grid(8 * 4 * 32 * 16, BT), BT, 0, stream>>>(W1, w1p);
    k_pack_w2<<<grid(4 * 32 * 16, BT), BT, 0, stream>>>(W2, w2p);

    // ---- GEMM1: h1 = xn @ W1 (WMMA f16 -> f32) ----
    // 3125 mtiles x 8 ntiles = 25000 waves, 8 waves/block
    k_gemm_wmma<<<MTILES, BT, 0, stream>>>(xnh, w1p, h1, 8, D1, D1);

    // ---- GAT layer 1 ----
    k_att1<<<grid((long)N_NODES * 32, BT), BT, 0, stream>>>(h1, att_s1, att_d1, al1, ar1);
    k_edge_logits1<<<grid((long)ETOT * H1, BT), BT, 0, stream>>>(ei, al1, ar1, m1, ex1);
    k_edge_exp1<<<grid((long)ETOT * H1, BT), BT, 0, stream>>>(ei, m1, ex1, den1);
    k_edge_aggr1<<<grid((long)ETOT * 32, BT), BT, 0, stream>>>(ei, ex1, den1, h1, out1);

    // ---- ELU + bias -> f16 (reuse xnh as x1h) ----
    k_elu_f16<<<grid((long)N_NODES * D1, BT), BT, 0, stream>>>(out1, b1, xnh);

    // ---- GEMM2: h2 = x1 @ W2 (WMMA, N padded to 16, store 8 cols) ----
    k_gemm_wmma<<<grid(MTILES, 8), BT, 0, stream>>>(xnh, w2p, h2, 1, C2, C2);

    // ---- GAT layer 2 ----
    k_att2<<<grid(N_NODES, BT), BT, 0, stream>>>(h2, att_s2, att_d2, al2, ar2);
    k_edge_logits2<<<grid(ETOT, BT), BT, 0, stream>>>(ei, al2, ar2, m2, ex2);
    k_edge_exp2<<<grid(ETOT, BT), BT, 0, stream>>>(ei, m2, ex2, den2);
    k_edge_aggr2<<<grid((long)ETOT * C2, BT), BT, 0, stream>>>(ei, ex2, den2, h2, out2);

    // ---- pool + log_softmax ----
    k_pool<<<grid((long)N_NODES * C2, BT), BT, 0, stream>>>(batch, out2, b2, pooled, cnt);
    k_logsoftmax<<<grid(G_GRAPHS, BT), BT, 0, stream>>>(pooled, cnt, y);
}
